// MultiHeadAttention_4209067950594
// MI455X (gfx1250) — matrix operand order repaired and checked by reading, not compiled, not run on hardware
//
#include <hip/hip_runtime.h>
#include <hip/hip_bf16.h>
#include <stdint.h>

// ---------------------------------------------------------------------------
// Linear multi-head attention (no softmax) for MI455X / gfx1250.
//   y = ((x Wq^T + bq)*s) @ [ (x Wk^T + bk)^T (x Wv^T + bv) ] per head,
//   out = y Wo^T + bo
// Exploits (qk^T)v == q(k^T v): attention cost drops 69 GF -> 8.6 GF and the
// 256MB [B,H,S,S] intermediate disappears.  All GEMMs run on
// V_WMMA_F32_16X16X32_BF16 (bf16 operands, f32 accumulate).  The shared B
// k-slab is double-buffered in LDS via GLOBAL_LOAD_ASYNC_TO_LDS_B128
// (ASYNCcnt), overlapping the DMA with the WMMA burst.
// ---------------------------------------------------------------------------

typedef __bf16 bf16_t;
typedef __attribute__((ext_vector_type(16))) __bf16 v16bf;
typedef __attribute__((ext_vector_type(8)))  __bf16 bf16x8;
typedef __attribute__((ext_vector_type(4)))  __bf16 bf16x4;
typedef __attribute__((ext_vector_type(8)))  float  v8f;

static constexpr int Bb = 2, Ss = 2048, Ee = 256, Hh = 8, HE = 2048;

// ---------------------------------------------------------------------------
// Async global->LDS copy (gfx1250 ASYNCcnt path) with sync fallback.
// Builtin signature (from hipcc diagnostic): arg0 = AS1 pointer to
// int __vector(4), arg1 = AS3 pointer, arg2/3 = imm offset / cpol.
// ---------------------------------------------------------------------------
#if defined(__HIP_DEVICE_COMPILE__) &&                                        \
    __has_builtin(__builtin_amdgcn_global_load_async_to_lds_b128) &&          \
    __has_builtin(__builtin_amdgcn_s_wait_asynccnt)
#define ASYNC_LDS 1
typedef int v4i_vs __attribute__((vector_size(16)));
typedef __attribute__((address_space(1))) v4i_vs* as1_b128_t;
typedef __attribute__((address_space(3))) v4i_vs* as3_b128_t;
__device__ __forceinline__ void async_b128(const bf16_t* g, bf16_t* l) {
  __builtin_amdgcn_global_load_async_to_lds_b128(
      (as1_b128_t)(uintptr_t)g,
      (as3_b128_t)(uint32_t)(uintptr_t)l,
      /*offset=*/0, /*cpol=*/0);
}
__device__ __forceinline__ void wait_async0() {
  __builtin_amdgcn_s_wait_asynccnt(0);
}
#else
__device__ __forceinline__ void async_b128(const bf16_t* g, bf16_t* l) {
  *(bf16x8*)l = *(const bf16x8*)g;   // global_load_b128 + ds_store_b128
}
__device__ __forceinline__ void wait_async0() {}
#endif

// ---------------------------------------------------------------------------
// f32 -> bf16 bulk convert (4 elements / thread)
// ---------------------------------------------------------------------------
__global__ void cvt_f32_bf16(const float* __restrict__ in,
                             bf16_t* __restrict__ out, int n) {
  int i = (blockIdx.x * blockDim.x + threadIdx.x) * 4;
  if (i < n) {
    float4 v = *(const float4*)(in + i);
    bf16x4 o;
    o[0] = (bf16_t)v.x; o[1] = (bf16_t)v.y;
    o[2] = (bf16_t)v.z; o[3] = (bf16_t)v.w;
    *(bf16x4*)(out + i) = o;
  }
}

// ---------------------------------------------------------------------------
// Generic batched NT GEMM on WMMA:  C[m,n] = (sum_k A[m,k]*B[n,k] + bias[n])*scale
// A: [M,K] row-major bf16 (lda), B: [N,K] row-major bf16 (ldb).
// Block = 256 threads = 8 waves (wave32), block tile 128x128,
// wave tile 32x64 -> 2x4 accumulators of 16x16, K step 32.
// B k-slab (128 rows x 32 k = 8KB bf16) is double-buffered in LDS via async
// copies and shared by all 8 waves; A fragments come straight from L2.
// MODE 0: bf16 store C[m*ldc+n]
// MODE 1: bf16 per-head transposed store  Kt[(n>>8)*E*S + (n&255)*S + m]
//         (contiguous along m -> packed 16B global_store_b128 per lane)
// MODE 2: f32 store  C[m*ldc+n]
// Batch offsets: z -> (zo=z/bdiv, zi=z%bdiv); ptr += zo*s?o + zi*s?i.
// All dims are exact multiples of the tiles for every call site -> no guards.
// ---------------------------------------------------------------------------
template <int MODE>
__global__ __launch_bounds__(256) void
wmma_nt_gemm(const bf16_t* __restrict__ Ag, const bf16_t* __restrict__ Bg,
             void* __restrict__ Cg, const float* __restrict__ bias,
             float scale, int Kd, int lda, int ldb, int ldc,
             long long sAo, long long sAi, long long sBo, long long sBi,
             long long sCo, long long sCi, int bdiv) {
  __shared__ __align__(16) bf16_t Bsh[2][128 * 32];  // 2 x 8 KB of 320 KB/WGP

  const int z  = blockIdx.z;
  const int zo = z / bdiv, zi = z % bdiv;
  Ag += zo * sAo + zi * sAi;
  Bg += zo * sBo + zi * sBi;
  const long long coff = zo * sCo + zi * sCi;

  const int tid  = threadIdx.x;
  const int wave = tid >> 5;           // 0..7
  const int lane = tid & 31;
  const int wm = wave >> 1;            // 0..3
  const int wn = wave & 1;             // 0..1
  const int l15 = lane & 15;
  const int lh  = lane >> 4;           // lane half
  const int m_wave  = blockIdx.y * 128 + wm * 32;
  const int n_block = blockIdx.x * 128;
  const int n_wave  = n_block + wn * 64;

  // cooperative B-slab fill: thread t -> row t>>1, k-half (t&1)*16 (32 bytes)
  const int fill_row = tid >> 1;        // 0..127
  const int fill_kh  = (tid & 1) * 16;  // 0 / 16

  auto fillB = [&](int buf, int k0) {
    const bf16_t* src =
        Bg + (long long)(n_block + fill_row) * ldb + k0 + fill_kh;
    bf16_t* dst = &Bsh[buf][fill_row * 32 + fill_kh];
    async_b128(src, dst);
    async_b128(src + 8, dst + 8);
  };

  v8f acc[2][4] = {};

  fillB(0, 0);  // prime buffer 0

  for (int k0 = 0; k0 < Kd; k0 += 32) {
    const int cur = (k0 >> 5) & 1;

    wait_async0();    // own async copies for Bsh[cur] landed in LDS
    __syncthreads();  // everyone's landed; prior reads of Bsh[cur] retired

    if (k0 + 32 < Kd) {
      fillB(cur ^ 1, k0 + 32);  // DMA next slab under the WMMA burst
      // prefetch next A k-tile (global_prefetch_b8)
      __builtin_prefetch(Ag + (long long)(m_wave + l15) * lda + k0 + 32, 0, 3);
    }

    // ---- A fragments (16x32 bf16) from global -----------------------------
    // lane holds row l15, K chunks {kb..kb+7, kb+16..kb+23}, kb = lh*8
    v16bf af[2];
#pragma unroll
    for (int i = 0; i < 2; ++i) {
      const bf16_t* ap =
          Ag + (long long)(m_wave + i * 16 + l15) * lda + k0 + lh * 8;
      bf16x8 lo = *(const bf16x8*)ap;
      bf16x8 hi = *(const bf16x8*)(ap + 16);
      af[i] = __builtin_shufflevector(lo, hi, 0, 1, 2, 3, 4, 5, 6, 7, 8, 9, 10,
                                      11, 12, 13, 14, 15);
    }

    // ---- B fragments (32x16 bf16) from LDS --------------------------------
    // lane = column l15, K chunks {kb..kb+7, kb+16..kb+23}, kb = lh*8 (2x ds_load_b128)
    v16bf bfr[4];
#pragma unroll
    for (int j = 0; j < 4; ++j) {
      const bf16_t* bp = &Bsh[cur][(wn * 64 + j * 16 + l15) * 32 + lh * 8];
      bf16x8 lo = *(const bf16x8*)bp;
      bf16x8 hi = *(const bf16x8*)(bp + 16);
      bfr[j] = __builtin_shufflevector(lo, hi, 0, 1, 2, 3, 4, 5, 6, 7, 8, 9, 10,
                                       11, 12, 13, 14, 15);
    }

#pragma unroll
    for (int i = 0; i < 2; ++i)
#pragma unroll
      for (int j = 0; j < 4; ++j)
        acc[i][j] = __builtin_amdgcn_wmma_f32_16x16x32_bf16(
            false, af[i], false, bfr[j], (short)0, acc[i][j], false, false);
  }

  // ---- Epilogue: C/D layout: lane column = l15, rows = lh*8 + r -----------
  if (MODE == 1) {
    // transposed store: memory contiguous along m; a lane's 8 rows are
    // consecutive m -> pack into one aligned 16B store per (i,j)
#pragma unroll
    for (int j = 0; j < 4; ++j) {
      const int n = n_wave + j * 16 + l15;
      const float bv = bias ? bias[n] : 0.0f;
      const long long nbase =
          coff + (long long)(n >> 8) * (Ee * Ss) + (long long)(n & (Ee - 1)) * Ss;
#pragma unroll
      for (int i = 0; i < 2; ++i) {
        bf16x8 pack;
#pragma unroll
        for (int r = 0; r < 8; ++r)
          pack[r] = (bf16_t)((acc[i][j][r] + bv) * scale);
        const int m0 = m_wave + i * 16 + lh * 8;
        *(bf16x8*)&((bf16_t*)Cg)[nbase + m0] = pack;
      }
    }
  } else {
#pragma unroll
    for (int j = 0; j < 4; ++j) {
      const int n = n_wave + j * 16 + l15;
      const float bv = bias ? bias[n] : 0.0f;
#pragma unroll
      for (int i = 0; i < 2; ++i) {
#pragma unroll
        for (int r = 0; r < 8; ++r) {
          const int m = m_wave + i * 16 + lh * 8 + r;
          const float val = (acc[i][j][r] + bv) * scale;
          if (MODE == 0)
            ((bf16_t*)Cg)[coff + (long long)m * ldc + n] = (bf16_t)val;
          else
            ((float*)Cg)[coff + (long long)m * ldc + n] = val;
        }
      }
    }
  }
}

// ---------------------------------------------------------------------------
extern "C" void kernel_launch(void* const* d_in, const int* in_sizes, int n_in,
                              void* d_out, int out_size, void* d_ws,
                              size_t ws_size, hipStream_t stream) {
  const float* x  = (const float*)d_in[0];
  const float* Wq = (const float*)d_in[1];
  const float* bq = (const float*)d_in[2];
  const float* Wk = (const float*)d_in[3];
  const float* bk = (const float*)d_in[4];
  const float* Wv = (const float*)d_in[5];
  const float* bv = (const float*)d_in[6];
  const float* Wo = (const float*)d_in[7];
  const float* bo = (const float*)d_in[8];
  float* out = (float*)d_out;

  // workspace layout (72 MB total), every region fully overwritten per call
  char* w = (char*)d_ws;
  bf16_t* Xb  = (bf16_t*)(w);                      //  2 MB  x  [4096,256]
  bf16_t* Wqb = (bf16_t*)(w + (2ull  << 20));      //  1 MB
  bf16_t* Wkb = (bf16_t*)(w + (3ull  << 20));      //  1 MB
  bf16_t* Wvb = (bf16_t*)(w + (4ull  << 20));      //  1 MB
  bf16_t* Wob = (bf16_t*)(w + (5ull  << 20));      //  1 MB
  bf16_t* Qb  = (bf16_t*)(w + (6ull  << 20));      // 16 MB  [b,s,h,e]
  bf16_t* Kt  = (bf16_t*)(w + (22ull << 20));      // 16 MB  [b,h,e,s]
  bf16_t* Vt  = (bf16_t*)(w + (38ull << 20));      // 16 MB  [b,h,e,s]
  bf16_t* KVt = (bf16_t*)(w + (54ull << 20));      //  2 MB  [bh,e2,e1]
  bf16_t* Yb  = (bf16_t*)(w + (56ull << 20));      // 16 MB  [b,s,h,e]

  auto cvt = [&](const float* src, bf16_t* dst, int n) {
    cvt_f32_bf16<<<dim3((n / 4 + 255) / 256), dim3(256), 0, stream>>>(src, dst, n);
  };
  cvt(x,  Xb,  Bb * Ss * Ee);
  cvt(Wq, Wqb, HE * Ee);
  cvt(Wk, Wkb, HE * Ee);
  cvt(Wv, Wvb, HE * Ee);
  cvt(Wo, Wob, Ee * HE);

  const float qscale = 0.0625f;  // E^-0.5 = 1/16

  // Q = (X Wq^T + bq) * scale  -> bf16 [4096, 2048]
  wmma_nt_gemm<0><<<dim3(HE / 128, (Bb * Ss) / 128, 1), dim3(256), 0, stream>>>(
      Xb, Wqb, Qb, bq, qscale, Ee, Ee, Ee, HE,
      0, 0, 0, 0, 0, 0, 1);

  // Kt[b,h,e,s] = (X Wk^T + bk)^T per head  (batched over b)
  wmma_nt_gemm<1><<<dim3(HE / 128, Ss / 128, Bb), dim3(256), 0, stream>>>(
      Xb, Wkb, Kt, bk, 1.0f, Ee, Ee, Ee, 0,
      (long long)Ss * Ee, 0, 0, 0, (long long)Hh * Ee * Ss, 0, 1);

  // Vt[b,h,e,s] likewise
  wmma_nt_gemm<1><<<dim3(HE / 128, Ss / 128, Bb), dim3(256), 0, stream>>>(
      Xb, Wvb, Vt, bv, 1.0f, Ee, Ee, Ee, 0,
      (long long)Ss * Ee, 0, 0, 0, (long long)Hh * Ee * Ss, 0, 1);

  // KVt[bh][e2,e1] = sum_s V[s,e2] K[s,e1]   (NT: A=Vt_h, B=Kt_h, K=S)
  wmma_nt_gemm<0><<<dim3(Ee / 128, Ee / 128, Bb * Hh), dim3(256), 0, stream>>>(
      Vt, Kt, KVt, nullptr, 1.0f, Ss, Ss, Ss, Ee,
      (long long)Ee * Ss, 0, (long long)Ee * Ss, 0, (long long)Ee * Ee, 0, 1);

  // Y[b,s,h,e2] = sum_e1 Q[b,s,h,e1] * KVt[bh][e2,e1]  (batched over b,h)
  wmma_nt_gemm<0><<<dim3(Ee / 128, Ss / 128, Bb * Hh), dim3(256), 0, stream>>>(
      Qb, KVt, Yb, nullptr, 1.0f, Ee, HE, Ee, HE,
      (long long)Ss * HE, Ee, (long long)Hh * Ee * Ee, (long long)Ee * Ee,
      (long long)Ss * HE, Ee, Hh);

  // out = Y Wo^T + bo  -> f32 [4096, 256]
  wmma_nt_gemm<2><<<dim3(Ee / 128, (Bb * Ss) / 128, 1), dim3(256), 0, stream>>>(
      Yb, Wob, out, bo, 1.0f, HE, HE, HE, Ee,
      0, 0, 0, 0, 0, 0, 1);
}
